// UnfoldConv1d_86947317940449
// MI455X (gfx1250) — matrix operand order repaired and checked by reading, not compiled, not run on hardware
//
#include <hip/hip_runtime.h>

typedef __bf16 bf16;
typedef __attribute__((ext_vector_type(2)))  float v2f;
typedef __attribute__((ext_vector_type(8)))  float v8f;
typedef __attribute__((ext_vector_type(8)))  bf16  v8bf;
typedef __attribute__((ext_vector_type(16))) bf16  v16bf;

namespace {
constexpr int C_IN  = 512;     // input channels
constexpr int T_LEN = 4096;    // time steps
constexpr int OUT_F = 512;     // output channels
constexpr int KF    = 1536;    // C_IN * kernel_size(3)
constexpr int B_SZ  = 8;       // batch
constexpr int PAD   = 8;       // left zero-pad rows in transposed x (>= 2, keeps 16B align)
constexpr int TP    = T_LEN + PAD;   // 4104 padded time rows

constexpr size_t W_ELEMS  = (size_t)OUT_F * KF;        // 786432
constexpr size_t XT_ELEMS = (size_t)B_SZ * TP * C_IN;  // 16,809,984
constexpr size_t OFF_WH = 0;
constexpr size_t OFF_WL = W_ELEMS * 2;                         // 1,572,864
constexpr size_t OFF_XH = 2 * W_ELEMS * 2;                     // 3,145,728
constexpr size_t OFF_XL = OFF_XH + XT_ELEMS * 2;               // 36,765,696
constexpr size_t WS_NEED = OFF_XL + XT_ELEMS * 2;              // 70,385,664 bytes
}

// ---------------------------------------------------------------------------
// Pre-pass 1: split W (f32, row-major (OUT, KF)) into bf16 hi/lo.
// ---------------------------------------------------------------------------
__global__ __launch_bounds__(256) void split_w_kernel(const float* __restrict__ W,
                                                      bf16* __restrict__ WH,
                                                      bf16* __restrict__ WL) {
    const size_t i = (size_t)blockIdx.x * 256 + threadIdx.x;
    if (i < W_ELEMS) {
        const float w = W[i];
        const bf16  h = (bf16)w;
        const bf16  l = (bf16)(w - (float)h);
        WH[i] = h;
        WL[i] = l;
    }
}

// ---------------------------------------------------------------------------
// Pre-pass 2: zero the PAD rows of the transposed/padded x arrays.
// Rows tp in [0, PAD) of each batch slab. 8 batches * PAD*C_IN elems.
// ---------------------------------------------------------------------------
__global__ __launch_bounds__(256) void zero_pad_rows_kernel(bf16* __restrict__ XH,
                                                            bf16* __restrict__ XL) {
    const int i = blockIdx.x * 256 + threadIdx.x;   // 0 .. B_SZ*PAD*C_IN-1 = 32767
    const int b = i >> 12;                          // PAD*C_IN = 4096 per batch
    const int r = i & 4095;
    const size_t idx = (size_t)b * TP * C_IN + r;
    XH[idx] = (bf16)0.0f;
    XL[idx] = (bf16)0.0f;
}

// ---------------------------------------------------------------------------
// Pre-pass 3: transpose + split x: (B, C, T) f32 -> XT[b][t+PAD][c] bf16 hi/lo.
// LDS-tiled 32x32 transpose so both global read and write are coalesced.
// grid: (T/32, C/32, B), block: (32, 8)
// ---------------------------------------------------------------------------
__global__ __launch_bounds__(256) void transpose_split_x_kernel(const float* __restrict__ X,
                                                                bf16* __restrict__ XH,
                                                                bf16* __restrict__ XL) {
    __shared__ float tile[32][33];
    const int b  = blockIdx.z;
    const int t0 = blockIdx.x * 32;
    const int c0 = blockIdx.y * 32;

    const float* xb = X + ((size_t)b * C_IN + c0) * T_LEN + t0;
    #pragma unroll
    for (int i = threadIdx.y; i < 32; i += 8)
        tile[i][threadIdx.x] = xb[(size_t)i * T_LEN + threadIdx.x];  // [c-local][t-local]
    __syncthreads();

    bf16* oh = XH + ((size_t)b * TP + t0 + PAD) * C_IN + c0;
    bf16* ol = XL + ((size_t)b * TP + t0 + PAD) * C_IN + c0;
    #pragma unroll
    for (int j = threadIdx.y; j < 32; j += 8) {
        const float v = tile[threadIdx.x][j];       // value at (c0+tx, t0+j)
        const bf16  h = (bf16)v;
        const bf16  l = (bf16)(v - (float)h);
        oh[(size_t)j * C_IN + threadIdx.x] = h;
        ol[(size_t)j * C_IN + threadIdx.x] = l;
    }
}

// ---------------------------------------------------------------------------
// Fragment assembly: 16 bf16 per lane from two aligned 16B chunks `gap` apart.
// ---------------------------------------------------------------------------
__device__ __forceinline__ v16bf ld_frag(const bf16* p, int gap) {
    const v8bf a = *(const v8bf*)p;
    const v8bf b = *(const v8bf*)(p + gap);
    return __builtin_shufflevector(a, b, 0, 1, 2, 3, 4, 5, 6, 7,
                                   8, 9, 10, 11, 12, 13, 14, 15);
}

__device__ __forceinline__ v8f wmma_bf16(v16bf a, v16bf b, v8f c) {
    return __builtin_amdgcn_wmma_f32_16x16x32_bf16(false, a, false, b,
                                                   (short)0, c, false, false);
}

// ---------------------------------------------------------------------------
// Main GEMM: Y[b,o,t] = sum_f W[o,f] * U[f,t] + bias[o], bf16x3 emulation on
// V_WMMA_F32_16X16X32_BF16 (f32 accumulate).
//   A = W tile 16x32 bf16 (hi & lo), per-lane chunks at kk+8*half, kk+16+8*half.
//   B = XT tile 32x16 bf16 (hi & lo), per-lane chunks at c+8*half, c+16+8*half,
//       row tp = t + (j-2) + PAD  (zero-pad rows make all taps unguarded).
// Wave tile: 4 M-tiles x 2 N-tiles = 64(o) x 32(t); 8 waves (2x4) -> 128x128/WG.
// Products per k-step: Ah*Bh + Ah*Bl + Al*Bh  (lo*lo dropped, ~2^-18 rel err).
// ---------------------------------------------------------------------------
__global__ __launch_bounds__(256)
void unfold_conv1d_wmma_bf16x3(const bf16* __restrict__ WH,
                               const bf16* __restrict__ WL,
                               const bf16* __restrict__ XH,
                               const bf16* __restrict__ XL,
                               const float* __restrict__ Bias,
                               float* __restrict__ Y) {
    const int lane = threadIdx.x & 31;
    const int wave = threadIdx.x >> 5;
    const int half = lane >> 4;
    const int n16  = lane & 15;

    const int nb = blockIdx.x & 31;          // 32 N-blocks of 128 cols
    const int mb = (blockIdx.x >> 5) & 3;    // 4  M-blocks of 128 rows
    const int bb = blockIdx.x >> 7;          // 8  batches

    const int mw = wave >> 2;                // 0..1
    const int nw = wave & 3;                 // 0..3

    const int mo = mb * 128 + mw * 64;       // wave M base (output channel)
    const int t0 = nb * 128 + nw * 32;       // wave N base (time)

    // A (weight) per-lane base pointers, per M-tile, hi and lo.
    const bf16* whp[4];
    const bf16* wlp[4];
    #pragma unroll
    for (int mi = 0; mi < 4; ++mi) {
        const size_t off = (size_t)(mo + mi * 16 + n16) * KF + 8 * half;
        whp[mi] = WH + off;
        wlp[mi] = WL + off;
    }

    v8f acc[4][2];
    const v8f vz = {0.f, 0.f, 0.f, 0.f, 0.f, 0.f, 0.f, 0.f};
    #pragma unroll
    for (int mi = 0; mi < 4; ++mi) {
        acc[mi][0] = vz;
        acc[mi][1] = vz;
    }

    // 3 causal taps, shift tsh = j-2 folded into the padded row index.
    #pragma unroll
    for (int j = 0; j < 3; ++j) {
        const int tsh = j - 2;
        const int kbase = j * C_IN;

        const bf16* bhp[2];
        const bf16* blp[2];
        #pragma unroll
        for (int ni = 0; ni < 2; ++ni) {
            const size_t off =
                ((size_t)bb * TP + (size_t)(t0 + ni * 16 + n16 + tsh + PAD)) * C_IN
                + 8 * half;
            bhp[ni] = XH + off;
            blp[ni] = XL + off;
        }

        #pragma unroll 2
        for (int c = 0; c < C_IN; c += 32) {
            const int kk = kbase + c;

            const v16bf bh0 = ld_frag(bhp[0] + c, 16);
            const v16bf bl0 = ld_frag(blp[0] + c, 16);
            const v16bf bh1 = ld_frag(bhp[1] + c, 16);
            const v16bf bl1 = ld_frag(blp[1] + c, 16);

            #pragma unroll
            for (int mi = 0; mi < 4; ++mi) {
                const v16bf ah = ld_frag(whp[mi] + kk, 16);
                const v16bf al = ld_frag(wlp[mi] + kk, 16);

                acc[mi][0] = wmma_bf16(ah, bh0, acc[mi][0]);
                acc[mi][0] = wmma_bf16(ah, bl0, acc[mi][0]);
                acc[mi][0] = wmma_bf16(al, bh0, acc[mi][0]);

                acc[mi][1] = wmma_bf16(ah, bh1, acc[mi][1]);
                acc[mi][1] = wmma_bf16(ah, bl1, acc[mi][1]);
                acc[mi][1] = wmma_bf16(al, bh1, acc[mi][1]);
            }
        }
    }

    // Epilogue: bias + coalesced stores. C/D layout: VGPR r -> M=r+8*half, N=n16.
    float* yb = Y + (size_t)bb * OUT_F * T_LEN;
    const int tt0 = t0 + n16;
    const int tt1 = t0 + 16 + n16;
    #pragma unroll
    for (int mi = 0; mi < 4; ++mi) {
        #pragma unroll
        for (int r = 0; r < 8; ++r) {
            const int o = mo + mi * 16 + 8 * half + r;
            const float bv = Bias[o];
            float* yrow = yb + (size_t)o * T_LEN;
            yrow[tt0] = acc[mi][0][r] + bv;
            yrow[tt1] = acc[mi][1][r] + bv;
        }
    }
}

// ---------------------------------------------------------------------------
// Fallback (no workspace): direct f32 GEMM on V_WMMA_F32_16X16X4_F32.
// ---------------------------------------------------------------------------
template <bool GUARD>
__device__ __forceinline__ void k_panel_f32(const float* __restrict__ xb,
                                            const float* const (&aptr)[4],
                                            int kkbase, int tsh,
                                            int tt0, int tt1, int half,
                                            v8f (&acc)[4][2]) {
    const int ta = tt0 + tsh;
    const int tb = tt1 + tsh;
    #pragma unroll 4
    for (int c = 0; c < C_IN; c += 4) {
        const float* xr = xb + (size_t)(c + 2 * half) * T_LEN;
        v2f b0, b1;
        if (GUARD) {
            b0.x = (ta >= 0) ? xr[ta] : 0.0f;
            b0.y = (ta >= 0) ? xr[T_LEN + ta] : 0.0f;
        } else {
            b0.x = xr[ta];
            b0.y = xr[T_LEN + ta];
        }
        b1.x = xr[tb];
        b1.y = xr[T_LEN + tb];

        const int kk = kkbase + c;
        #pragma unroll
        for (int mi = 0; mi < 4; ++mi) {
            v2f a = *(const v2f*)(aptr[mi] + kk);
            acc[mi][0] = __builtin_amdgcn_wmma_f32_16x16x4_f32(
                false, a, false, b0, (short)0, acc[mi][0], false, false);
            acc[mi][1] = __builtin_amdgcn_wmma_f32_16x16x4_f32(
                false, a, false, b1, (short)0, acc[mi][1], false, false);
        }
    }
}

__global__ __launch_bounds__(256)
void unfold_conv1d_wmma_f32(const float* __restrict__ X,
                            const float* __restrict__ Wt,
                            const float* __restrict__ Bias,
                            float* __restrict__ Y) {
    const int lane = threadIdx.x & 31;
    const int wave = threadIdx.x >> 5;
    const int half = lane >> 4;
    const int n16  = lane & 15;

    const int nb = blockIdx.x & 31;
    const int mb = (blockIdx.x >> 5) & 3;
    const int bb = blockIdx.x >> 7;
    const int mw = wave >> 2;
    const int nw = wave & 3;
    const int mo = mb * 128 + mw * 64;
    const int t0 = nb * 128 + nw * 32;

    const float* aptr[4];
    #pragma unroll
    for (int mi = 0; mi < 4; ++mi)
        aptr[mi] = Wt + (size_t)(mo + mi * 16 + n16) * KF + 2 * half;

    const float* xb = X + (size_t)bb * C_IN * T_LEN;
    const int tt0 = t0 + n16;
    const int tt1 = t0 + 16 + n16;
    const bool edge = (t0 == 0);

    v8f acc[4][2];
    const v8f vz = {0.f, 0.f, 0.f, 0.f, 0.f, 0.f, 0.f, 0.f};
    #pragma unroll
    for (int mi = 0; mi < 4; ++mi) {
        acc[mi][0] = vz;
        acc[mi][1] = vz;
    }

    #pragma unroll
    for (int j = 0; j < 3; ++j) {
        const int tsh = j - 2;
        if (edge && j != 2)
            k_panel_f32<true >(xb, aptr, j * C_IN, tsh, tt0, tt1, half, acc);
        else
            k_panel_f32<false>(xb, aptr, j * C_IN, tsh, tt0, tt1, half, acc);
    }

    float* yb = Y + (size_t)bb * OUT_F * T_LEN;
    #pragma unroll
    for (int mi = 0; mi < 4; ++mi) {
        #pragma unroll
        for (int r = 0; r < 8; ++r) {
            const int o = mo + mi * 16 + 8 * half + r;
            const float bv = Bias[o];
            float* yrow = yb + (size_t)o * T_LEN;
            yrow[tt0] = acc[mi][0][r] + bv;
            yrow[tt1] = acc[mi][1][r] + bv;
        }
    }
}

// ---------------------------------------------------------------------------
extern "C" void kernel_launch(void* const* d_in, const int* in_sizes, int n_in,
                              void* d_out, int out_size, void* d_ws, size_t ws_size,
                              hipStream_t stream) {
    const float* X  = (const float*)d_in[0];   // (8, 512, 4096) f32
    const float* W  = (const float*)d_in[1];   // (512, 1536)    f32
    const float* Bv = (const float*)d_in[2];   // (512)          f32
    float* Y = (float*)d_out;                  // (8, 512, 4096) f32

    if (ws_size >= WS_NEED && d_ws != nullptr) {
        char* ws = (char*)d_ws;
        bf16* WH = (bf16*)(ws + OFF_WH);
        bf16* WL = (bf16*)(ws + OFF_WL);
        bf16* XH = (bf16*)(ws + OFF_XH);
        bf16* XL = (bf16*)(ws + OFF_XL);

        // Pre-passes (stream-ordered): split W, zero pad rows, transpose+split x.
        hipLaunchKernelGGL(split_w_kernel, dim3((W_ELEMS + 255) / 256), dim3(256),
                           0, stream, W, WH, WL);
        hipLaunchKernelGGL(zero_pad_rows_kernel, dim3((B_SZ * PAD * C_IN) / 256),
                           dim3(256), 0, stream, XH, XL);
        hipLaunchKernelGGL(transpose_split_x_kernel,
                           dim3(T_LEN / 32, C_IN / 32, B_SZ), dim3(32, 8),
                           0, stream, X, XH, XL);

        // Main GEMM: batches * M-blocks * N-blocks = 8*4*32 = 1024 workgroups.
        hipLaunchKernelGGL(unfold_conv1d_wmma_bf16x3, dim3(32 * 4 * 8), dim3(256),
                           0, stream, WH, WL, XH, XL, Bv, Y);
    } else {
        hipLaunchKernelGGL(unfold_conv1d_wmma_f32, dim3(32 * 4 * 8), dim3(256),
                           0, stream, X, W, Bv, Y);
    }
}
